// NTXenLoss_Mixed_23742579213070
// MI455X (gfx1250) — matrix operand order repaired and checked by reading, not compiled, not run on hardware
//
#include <hip/hip_runtime.h>
#include <hip/hip_bf16.h>
#include <math.h>

// ---------------------------------------------------------------------------
// NTXent-style NNCLR loss for MI455X (gfx1250, wave32, WMMA).
// N=8192, D=128. 7 full NxN GEMMs via v_wmma_f32_16x16x32_f16, LDS-staged.
// ---------------------------------------------------------------------------

#define GN 8192
#define GD 128
#define GINF 1.0e8f
#define GINVT 10.0f          // 1/TEMP
#define NEGBIG -3.0e38f
#define LDA 136              // padded LDS row stride (halves): 16B skew per row
#define LDB 136

typedef __attribute__((ext_vector_type(16))) _Float16 v16h;
typedef __attribute__((ext_vector_type(8)))  _Float16 v8h;
typedef __attribute__((ext_vector_type(8)))  float    v8f;

// ---------------- helpers ---------------------------------------------------

__device__ inline unsigned f2key(float x) {
  unsigned u = __float_as_uint(x);
  return (u & 0x80000000u) ? ~u : (u | 0x80000000u);
}
__device__ inline float key2f(unsigned k) {
  unsigned u = (k & 0x80000000u) ? (k & 0x7fffffffu) : ~k;
  return __uint_as_float(u);
}
__device__ inline void lse_upd(float& m, float& s, float v) {
  if (v > m) { s = s * __expf(m - v) + 1.0f; m = v; }
  else       { s += __expf(v - m); }
}
// block-wide online-logsumexp merge over 256 threads; valid in all threads
__device__ inline float block_lse(float m, float s) {
  __shared__ float sm[256], ss[256];
  int t = threadIdx.x;
  sm[t] = m; ss[t] = s; __syncthreads();
  for (int off = 128; off > 0; off >>= 1) {
    if (t < off) {
      float m2 = sm[t + off], s2 = ss[t + off];
      float mn = fmaxf(sm[t], m2);
      ss[t] = ss[t] * __expf(sm[t] - mn) + s2 * __expf(m2 - mn);
      sm[t] = mn;
    }
    __syncthreads();
  }
  return sm[0] + __logf(ss[0]);
}
__device__ inline v16h frag16(const _Float16* p0, const _Float16* p1) {
  v8h lo = *(const v8h*)p0;
  v8h hi = *(const v8h*)p1;
  return __builtin_shufflevector(lo, hi,
                                 0, 1, 2, 3, 4, 5, 6, 7,
                                 8, 9, 10, 11, 12, 13, 14, 15);
}

// ---------------- kernel: normalize (f32 + f16 copies) ----------------------

__global__ __launch_bounds__(128)
void normalize_k(const float* __restrict__ zi, const float* __restrict__ zj,
                 float* __restrict__ zin, float* __restrict__ zjn,
                 _Float16* __restrict__ zih, _Float16* __restrict__ zjh) {
  const int row = blockIdx.x, t = threadIdx.x;
  const float* src = blockIdx.y ? zj : zi;
  float*      dn  = blockIdx.y ? zjn : zin;
  _Float16*   dh  = blockIdx.y ? zjh : zih;
  float x = src[(size_t)row * GD + t];
  __shared__ float red[128];
  red[t] = x * x; __syncthreads();
  for (int off = 64; off > 0; off >>= 1) {
    if (t < off) red[t] += red[t + off];
    __syncthreads();
  }
  float nrm = fmaxf(sqrtf(red[0]), 1e-12f);
  float v = x / nrm;
  dn[(size_t)row * GD + t] = v;
  dh[(size_t)row * GD + t] = (_Float16)v;
}

// ---------------- kernel: WMMA GEMM  C = scale * A @ B^T --------------------
// LDS-staged: block tile 128x64, full K=128 staged once (A 128x128 f16,
// B 64x128 f16, ~51KB of the WGP's 320KB LDS). A rows optionally gathered
// through a_idx at stage time. Each wave computes a 16x64 strip: A fragment
// reused across 4 WMMA tiles; 16 v_wmma_f32_16x16x32_f16 per wave.

__global__ __launch_bounds__(256)
void wmma_gemm_nt(const _Float16* __restrict__ A, const _Float16* __restrict__ B,
                  const int* __restrict__ a_idx, float* __restrict__ C, float scale) {
  __shared__ _Float16 As[128 * LDA];
  __shared__ _Float16 Bs[64 * LDB];

  const int t    = threadIdx.x;
  const int lane = t & 31;
  const int wave = t >> 5;
  const int hs   = lane >> 4;        // lane half select
  const int l15  = lane & 15;
  const int m0   = blockIdx.y * 128;
  const int n0   = blockIdx.x * 64;

  // --- cooperative global -> LDS stage (coalesced b128 chunks) ---
  {
    const int chunk = t & 15;        // 16 x 8-half chunks per 128-half row
    const int row0  = t >> 4;        // 16 rows per sweep
    for (int r = row0; r < 128; r += 16) {
      int grow = m0 + r;
      if (a_idx) grow = a_idx[grow];
      *(v8h*)(As + r * LDA + chunk * 8) =
          *(const v8h*)(A + (size_t)grow * GD + chunk * 8);
    }
    for (int r = row0; r < 64; r += 16) {
      *(v8h*)(Bs + r * LDB + chunk * 8) =
          *(const v8h*)(B + (size_t)(n0 + r) * GD + chunk * 8);
    }
  }
  __syncthreads();

  // --- compute from LDS ---
  const _Float16* ap = As + (wave * 16 + l15) * LDA;     // A row for this lane
  const _Float16* bp = Bs + l15 * LDB + hs * 16;         // B base col for lane

  v8f acc0 = {}, acc1 = {}, acc2 = {}, acc3 = {};
#pragma unroll
  for (int kb = 0; kb < GD; kb += 32) {
    // A fragment (16x32 f16): lane half hs holds contiguous K = hs*16+[0..15]
    v16h af = frag16(ap + kb + hs * 16, ap + kb + hs * 16 + 8);
    // B fragments (32x16 f16): lane = column, contiguous 16 K at hs*16
    v16h b0 = frag16(bp + kb,            bp + kb + 8);
    v16h b1 = frag16(bp + 16 * LDB + kb, bp + 16 * LDB + kb + 8);
    v16h b2 = frag16(bp + 32 * LDB + kb, bp + 32 * LDB + kb + 8);
    v16h b3 = frag16(bp + 48 * LDB + kb, bp + 48 * LDB + kb + 8);
    acc0 = __builtin_amdgcn_wmma_f32_16x16x32_f16(false, af, false, b0, (short)0, acc0, false, false);
    acc1 = __builtin_amdgcn_wmma_f32_16x16x32_f16(false, af, false, b1, (short)0, acc1, false, false);
    acc2 = __builtin_amdgcn_wmma_f32_16x16x32_f16(false, af, false, b2, (short)0, acc2, false, false);
    acc3 = __builtin_amdgcn_wmma_f32_16x16x32_f16(false, af, false, b3, (short)0, acc3, false, false);
  }

  // C/D layout: VGPR g, lane l -> row base + g + hs*8, col n0 + (l&15)
  const size_t rbase = (size_t)(m0 + wave * 16 + hs * 8);
#pragma unroll
  for (int g = 0; g < 8; ++g) {
    size_t ro = (rbase + g) * (size_t)GN + n0 + l15;
    C[ro +  0] = acc0[g] * scale;
    C[ro + 16] = acc1[g] * scale;
    C[ro + 32] = acc2[g] * scale;
    C[ro + 48] = acc3[g] * scale;
  }
}

// ---------------- kernel: row argmax (diag excluded) ------------------------

__global__ __launch_bounds__(256)
void row_argmax_k(const float* __restrict__ S, int* __restrict__ idx) {
  const int r = blockIdx.x, t = threadIdx.x;
  float bv = NEGBIG; int bi = GN;
  for (int j = t; j < GN; j += 256) {
    if (j == r) continue;
    float v = S[(size_t)r * GN + j];
    if (v > bv || (v == bv && j < bi)) { bv = v; bi = j; }
  }
  __shared__ float vs[256]; __shared__ int is[256];
  vs[t] = bv; is[t] = bi; __syncthreads();
  for (int off = 128; off > 0; off >>= 1) {
    if (t < off) {
      if (vs[t + off] > vs[t] || (vs[t + off] == vs[t] && is[t + off] < is[t])) {
        vs[t] = vs[t + off]; is[t] = is[t + off];
      }
    }
    __syncthreads();
  }
  if (t == 0) idx[r] = is[0];
}

// ---------------- kernel: column argmax (diag excluded, coalesced) ----------

__global__ __launch_bounds__(256)
void col_argmax_k(const float* __restrict__ S, int* __restrict__ idx) {
  const int c = blockIdx.x * 256 + threadIdx.x;
  float bv = NEGBIG; int bi = 0;
  for (int i = 0; i < GN; ++i) {
    if (i == c) continue;
    float v = S[(size_t)i * GN + c];
    if (v > bv) { bv = v; bi = i; }   // strict > keeps first max (jnp.argmax)
  }
  idx[c] = bi;
}

// ---------------- radix-select quantile (dual rank, 3 passes) ---------------
// state layout (6 u32 / selection): prefix0, prefix1, mask0, mask1, rem0, rem1

__global__ __launch_bounds__(256)
void select_init_k(unsigned* __restrict__ hist, unsigned* __restrict__ states) {
  for (int i = threadIdx.x; i < 4096; i += 256) hist[i] = 0;
  if (threadIdx.x < 3) {
    unsigned* st = states + threadIdx.x * 6;
    st[0] = 0; st[1] = 0; st[2] = 0; st[3] = 0;
    st[4] = 50331647u;   // k   = floor(0.75*(N*N-1))
    st[5] = 50331648u;   // k+1
  }
}

__global__ __launch_bounds__(256)
void hist_k(const float* __restrict__ buf, unsigned* __restrict__ hist,
            const unsigned* __restrict__ st, int shift, int nbins) {
  __shared__ unsigned lh[4096];
  for (int i = threadIdx.x; i < 4096; i += 256) lh[i] = 0;
  __syncthreads();
  const unsigned p0 = st[0], p1 = st[1], m0 = st[2], m1 = st[3];
  const size_t total = (size_t)GN * GN;
  const size_t stride = (size_t)gridDim.x * 256;
  for (size_t i = (size_t)blockIdx.x * 256 + threadIdx.x; i < total; i += stride) {
    unsigned k = f2key(buf[i]);
    unsigned b = (k >> shift) & (unsigned)(nbins - 1);
    if ((k & m0) == p0) atomicAdd(&lh[b], 1u);
    if ((k & m1) == p1) atomicAdd(&lh[2048 + b], 1u);
  }
  __syncthreads();
  for (int i = threadIdx.x; i < 2048; i += 256) {
    if (lh[i])        atomicAdd(&hist[i],        lh[i]);
    if (lh[2048 + i]) atomicAdd(&hist[2048 + i], lh[2048 + i]);
  }
}

__global__ __launch_bounds__(256)
void refine_k(unsigned* __restrict__ hist, unsigned* __restrict__ st,
              int shift, int nbins, float* __restrict__ qout) {
  if (threadIdx.x == 0) {
    for (int a = 0; a < 2; ++a) {
      unsigned rem = st[4 + a];
      const unsigned* h = hist + a * 2048;
      unsigned b = 0;
      for (; b < (unsigned)nbins; ++b) {
        unsigned c = h[b];
        if (rem < c) break;
        rem -= c;
      }
      if (b >= (unsigned)nbins) b = (unsigned)nbins - 1;
      st[a]     |= b << shift;
      st[2 + a] |= (unsigned)(nbins - 1) << shift;
      st[4 + a]  = rem;
    }
    if (shift == 0) {
      // jnp.quantile linear interpolation: frac = 0.25
      *qout = 0.75f * key2f(st[0]) + 0.25f * key2f(st[1]);
    }
  }
  __syncthreads();
  for (int i = threadIdx.x; i < 4096; i += 256) hist[i] = 0;
}

// ---------------- kernel: in-place mask sim_zij -> sim_zij_mod --------------

__global__ __launch_bounds__(256)
void mask_zij_k(float* __restrict__ S, const float* __restrict__ qp) {
  const float q = *qp;
  const int r = blockIdx.y;
  const int c = blockIdx.x * 256 + threadIdx.x;
  size_t o = (size_t)r * GN + c;
  float v = S[o];
  if (r != c) S[o] = (v > q) ? -GINF : v;
}

// ---------------- kernel: WHN loss_i per-row LSE ----------------------------

__global__ __launch_bounds__(256)
void whn_row_i_k(const float* __restrict__ Smod, const float* __restrict__ Sii,
                 const float* __restrict__ qiip, float* __restrict__ outrow) {
  const int r = blockIdx.x, t = threadIdx.x;
  const float qii = *qiip;
  const float* r1 = Smod + (size_t)r * GN;
  const float* r2 = Sii + (size_t)r * GN;
  float m = NEGBIG, s = 0.0f;
  for (int j = t; j < GN; j += 256) {
    lse_upd(m, s, r1[j]);
    float w = r2[j];
    w = (j == r) ? (w - GINF) : ((w > qii) ? -GINF : w);
    lse_upd(m, s, w);
  }
  float lse = block_lse(m, s);
  if (t == 0) outrow[r] = lse - r1[r];
}

// ---------------- kernel: WHN loss_j per-column LSE (coalesced) -------------

__global__ __launch_bounds__(256)
void whn_col_j_k(const float* __restrict__ Smod, const float* __restrict__ Sjj,
                 const float* __restrict__ qjjp, float* __restrict__ outrow) {
  const int c = blockIdx.x * 256 + threadIdx.x;
  const float qjj = *qjjp;
  float m = NEGBIG, s = 0.0f;
  for (int i = 0; i < GN; ++i) {
    lse_upd(m, s, Smod[(size_t)i * GN + c]);
    float w = Sjj[(size_t)i * GN + c];   // Sjj symmetric: column c == row c
    w = (i == c) ? (w - GINF) : ((w > qjj) ? -GINF : w);
    lse_upd(m, s, w);
  }
  outrow[c] = (m + __logf(s)) - Smod[(size_t)c * GN + c];
}

// ---------------- kernel: NN loss per-row LSE -------------------------------

__global__ __launch_bounds__(256)
void nn_row_k(const float* __restrict__ Sfirst, const float* __restrict__ Ssecond,
              const int* __restrict__ midx, float* __restrict__ outrow) {
  const int r = blockIdx.x, t = threadIdx.x;
  const int mi = midx[r];
  const float* r1 = Sfirst + (size_t)r * GN;
  const float* r2 = Ssecond + (size_t)r * GN;
  float m = NEGBIG, s = 0.0f;
  for (int j = t; j < GN; j += 256) {
    float v = r1[j];
    if (j == mi) v = -GINF;            // NN itself removed
    lse_upd(m, s, v);
    float w = r2[j];
    if (j == r) w -= GINF;             // diag_inf on same-view block
    lse_upd(m, s, w);
  }
  float lse = block_lse(m, s);
  if (t == 0) outrow[r] = lse - r1[r]; // label logit = z_nn[r] . z_other[r]
}

// ---------------- kernel: deterministic final reduction ---------------------

__global__ __launch_bounds__(256)
void final_k(const float* __restrict__ a, const float* __restrict__ b,
             const float* __restrict__ c, const float* __restrict__ d,
             float* __restrict__ out0) {
  const int t = threadIdx.x;
  float s = 0.0f;
  for (int i = t; i < GN; i += 256) s += a[i] + b[i] + c[i] + d[i];
  __shared__ float red[256];
  red[t] = s; __syncthreads();
  for (int off = 128; off > 0; off >>= 1) {
    if (t < off) red[t] += red[t + off];
    __syncthreads();
  }
  if (t == 0) *out0 = 0.5f * (red[0] / (float)GN);
}

__global__ __launch_bounds__(256)
void pairs_k(float* __restrict__ out) {
  const int i = blockIdx.x * 256 + threadIdx.x;
  out[i] = (float)i;
}

// ---------------------------------------------------------------------------

extern "C" void kernel_launch(void* const* d_in, const int* in_sizes, int n_in,
                              void* d_out, int out_size, void* d_ws, size_t ws_size,
                              hipStream_t stream) {
  (void)in_sizes; (void)n_in; (void)out_size; (void)ws_size;
  const float* z_i = (const float*)d_in[0];
  const float* z_j = (const float*)d_in[1];

  const size_t ND = (size_t)GN * GD;
  const size_t NN = (size_t)GN * GN;

  float*     zi_n  = (float*)d_ws;
  float*     zj_n  = zi_n + ND;
  _Float16*  zi_h  = (_Float16*)(zj_n + ND);
  _Float16*  zj_h  = zi_h + ND;
  float*     bufA  = (float*)(zj_h + ND);   // sim_zii, later sim_nn_*
  float*     bufB  = bufA + NN;             // sim_zjj, later sim_nn_*
  float*     rw_i  = bufB + NN;             // whn loss_i per row
  float*     rw_j  = rw_i + GN;
  float*     rn_i  = rw_j + GN;             // nn loss_i per row
  float*     rn_j  = rn_i + GN;
  int*       idx_ij = (int*)(rn_j + GN);
  int*       idx_ji = idx_ij + GN;
  unsigned*  hist   = (unsigned*)(idx_ji + GN);   // 4096
  unsigned*  states = hist + 4096;                // 3 * 6
  float*     qvals  = (float*)(states + 18);      // q_ij, q_ii, q_jj

  float* out  = (float*)d_out;
  float* sim  = out + 1;                    // sim_zij -> sim_zij_mod in place

  const dim3 gemm_grid(GN / 64, GN / 128);

  // 0) select state init
  select_init_k<<<1, 256, 0, stream>>>(hist, states);
  // 1) normalize
  normalize_k<<<dim3(GN, 2), 128, 0, stream>>>(z_i, z_j, zi_n, zj_n, zi_h, zj_h);
  // 2) similarity GEMMs
  wmma_gemm_nt<<<gemm_grid, 256, 0, stream>>>(zi_h, zj_h, nullptr, sim,  GINVT); // sim_zij
  wmma_gemm_nt<<<gemm_grid, 256, 0, stream>>>(zi_h, zi_h, nullptr, bufA, GINVT); // sim_zii
  wmma_gemm_nt<<<gemm_grid, 256, 0, stream>>>(zj_h, zj_h, nullptr, bufB, GINVT); // sim_zjj
  // 3) nearest-neighbor indices (diag excluded)
  row_argmax_k<<<GN, 256, 0, stream>>>(sim, idx_ij);
  col_argmax_k<<<GN / 256, 256, 0, stream>>>(sim, idx_ji);
  // 4) exact 0.75-quantiles via dual-rank radix select (3 passes, 32 bits)
  const int   shifts[3] = {21, 10, 0};
  const int   nbins[3]  = {2048, 2048, 1024};
  const float* qsrc[3]  = {sim, bufA, bufB};
  for (int sidx = 0; sidx < 3; ++sidx) {
    for (int p = 0; p < 3; ++p) {
      hist_k<<<2048, 256, 0, stream>>>(qsrc[sidx], hist, states + sidx * 6,
                                       shifts[p], nbins[p]);
      refine_k<<<1, 256, 0, stream>>>(hist, states + sidx * 6,
                                      shifts[p], nbins[p], qvals + sidx);
    }
  }
  // 5) sim_zij -> sim_zij_mod in place (diag kept; off-diag > q -> -INF)
  mask_zij_k<<<dim3(GN / 256, GN), 256, 0, stream>>>(sim, qvals + 0);
  // 6) WHN cross-entropies
  whn_row_i_k<<<GN, 256, 0, stream>>>(sim, bufA, qvals + 1, rw_i);
  whn_col_j_k<<<GN / 256, 256, 0, stream>>>(sim, bufB, qvals + 2, rw_j);
  // 7) NN loss i-side: z_nn_i = zj[idx_ij]
  wmma_gemm_nt<<<gemm_grid, 256, 0, stream>>>(zj_h, zj_h, idx_ij, bufA, GINVT); // sim_nn_zij
  wmma_gemm_nt<<<gemm_grid, 256, 0, stream>>>(zj_h, zi_h, idx_ij, bufB, GINVT); // sim_nn_zii
  nn_row_k<<<GN, 256, 0, stream>>>(bufA, bufB, idx_ij, rn_i);
  // 8) NN loss j-side: z_nn_j = zi[idx_ji]
  wmma_gemm_nt<<<gemm_grid, 256, 0, stream>>>(zi_h, zi_h, idx_ji, bufA, GINVT); // sim_nn_zji
  wmma_gemm_nt<<<gemm_grid, 256, 0, stream>>>(zi_h, zj_h, idx_ji, bufB, GINVT); // sim_nn_zjj
  nn_row_k<<<GN, 256, 0, stream>>>(bufA, bufB, idx_ji, rn_j);
  // 9) final scalar + correct_pairs
  final_k<<<1, 256, 0, stream>>>(rn_i, rn_j, rw_i, rw_j, out);
  pairs_k<<<GN / 256, 256, 0, stream>>>(out + 1 + NN);
}
